// SelfAttention_13314398617954
// MI455X (gfx1250) — matrix operand order repaired and checked by reading, not compiled, not run on hardware
//
#include <hip/hip_runtime.h>

typedef __attribute__((ext_vector_type(16))) __bf16 bf16x16;
typedef __attribute__((ext_vector_type(8)))  float  floatx8;

#define B_SZ   4
#define T_SZ   4096
#define D_SZ   64
#define TQ     16            // queries per wave
#define WAVES  4
#define QBLK   (TQ * WAVES)  // 64 queries per block
#define KC     64            // keys per chunk

__device__ __forceinline__ unsigned short f2bf(float f) {
    unsigned int u = __float_as_uint(f);
    u += 0x7FFFu + ((u >> 16) & 1u);   // round-to-nearest-even
    return (unsigned short)(u >> 16);
}

// Pack two f32 -> {bf16(hi), bf16(lo)} in one dword: bias-add + v_perm_b32.
__device__ __forceinline__ unsigned int pack2_bf16(float hi, float lo) {
    unsigned int a = __float_as_uint(lo) + 0x8000u;  // round-half-up
    unsigned int b = __float_as_uint(hi) + 0x8000u;
    return __builtin_amdgcn_perm(b, a, 0x07060302u); // {b[31:16], a[31:16]}
}

// Build a v16 bf16 fragment from two 16-byte LDS regions.
__device__ __forceinline__ bf16x16 ld_frag(const unsigned short* p0,
                                           const unsigned short* p1) {
    union { bf16x16 v; uint4 q[2]; } u;
    u.q[0] = *(const uint4*)p0;
    u.q[1] = *(const uint4*)p1;
    return u.v;
}

// xor-max over 16-lane halves via ds_swizzle (imm = (xor_mask<<10) | 0x1f)
#define SWZ_MAXF(x, imm)                                                       \
    (x) = fmaxf((x), __int_as_float(                                           \
              __builtin_amdgcn_ds_swizzle(__float_as_int(x), (imm))))

__global__ __launch_bounds__(128)
void fa_bf16_wmma_kernel(const float* __restrict__ Q,
                         const float* __restrict__ Km,
                         const float* __restrict__ Vm,
                         const float* __restrict__ mask,
                         float* __restrict__ out) {
    __shared__ unsigned short sK[KC * D_SZ];          // [key][d]  bf16
    __shared__ unsigned short sVT[D_SZ * KC];         // [d][key]  bf16 (transposed V)
    __shared__ unsigned short sP[WAVES][TQ * KC];     // per-wave P tile [row][key]

    const int tid  = threadIdx.x;
    const int wave = tid >> 5;
    const int lane = tid & 31;
    const int h    = lane >> 4;     // half of wave (0: lanes 0-15, 1: lanes 16-31)
    const int ln   = lane & 15;

    const int blocksPerBatch = T_SZ / QBLK;           // 64
    const int batch = blockIdx.x / blocksPerBatch;
    const int q0    = (blockIdx.x % blocksPerBatch) * QBLK + wave * TQ;

    // ---- Q tile as two bf16 A-fragments (K-steps over D: [0,32) and [32,64)) ----
    bf16x16 aQ[2];
    {
        const float* qrow = Q + ((size_t)batch * T_SZ + (q0 + ln)) * D_SZ;
        #pragma unroll
        for (int s = 0; s < 2; ++s) {
            union { bf16x16 v; unsigned short u[16]; } fr;
            const int d0 = 32 * s + 8 * h;        // elements 0..7  -> K = d0..d0+7
            const int d1 = 32 * s + 16 + 8 * h;   // elements 8..15 -> K = d1..d1+7
            #pragma unroll
            for (int e = 0; e < 8; ++e) fr.u[e]     = f2bf(qrow[d0 + e]);
            #pragma unroll
            for (int e = 0; e < 8; ++e) fr.u[8 + e] = f2bf(qrow[d1 + e]);
            aQ[s] = fr.v;
        }
    }

    // ---- all-ones bf16 B fragment (for row-sum-by-WMMA) ----
    bf16x16 bOnes;
    {
        union { bf16x16 v; unsigned short u[16]; } on;
        #pragma unroll
        for (int e = 0; e < 16; ++e) on.u[e] = 0x3F80;   // bf16 1.0
        bOnes = on.v;
    }

    // ---- Online-softmax state (C-fragment row mapping: row = r + 8*h) ----
    float m_r[8];
    floatx8 o[4], oL;   // oL accumulates row sums (every lane of the row identical)
    #pragma unroll
    for (int r = 0; r < 8; ++r) m_r[r] = -1e30f;
    #pragma unroll
    for (int t = 0; t < 4; ++t)
        #pragma unroll
        for (int e = 0; e < 8; ++e) o[t][e] = 0.0f;
    #pragma unroll
    for (int e = 0; e < 8; ++e) oL[e] = 0.0f;

    const float scale = 1.0f / (float)D_SZ;
    const float* mrow = mask + ((size_t)batch * T_SZ + q0) * T_SZ;
    const int NCHUNK = T_SZ / KC;

    for (int kb = 0; kb < NCHUNK; ++kb) {
        const int k0 = kb * KC;

        __syncthreads();   // protect previous iteration's LDS reads

        // ---- Stage K chunk (row-major bf16) and V chunk (transposed bf16) ----
        {
            const float* kbase = Km + ((size_t)batch * T_SZ + k0) * D_SZ;
            const float* vbase = Vm + ((size_t)batch * T_SZ + k0) * D_SZ;
            #pragma unroll
            for (int i = 0; i < 8; ++i) {
                const int idx = tid + i * 128;        // 1024 float4 units = 64x64 floats
                const int key = idx >> 4;
                const int d4  = (idx & 15) << 2;
                const float4 f = *(const float4*)(kbase + key * D_SZ + d4);
                *(uint2*)&sK[key * D_SZ + d4] =
                    make_uint2(pack2_bf16(f.y, f.x), pack2_bf16(f.w, f.z));

                const float4 v = *(const float4*)(vbase + key * D_SZ + d4);
                sVT[(d4 + 0) * KC + key] = (unsigned short)((__float_as_uint(v.x) + 0x8000u) >> 16);
                sVT[(d4 + 1) * KC + key] = (unsigned short)((__float_as_uint(v.y) + 0x8000u) >> 16);
                sVT[(d4 + 2) * KC + key] = (unsigned short)((__float_as_uint(v.z) + 0x8000u) >> 16);
                sVT[(d4 + 3) * KC + key] = (unsigned short)((__float_as_uint(v.w) + 0x8000u) >> 16);
            }
        }
        __syncthreads();

        // ---- prefetch next chunk's mask tile (16 rows x 256B = 32 cachelines) ----
        if (kb + 1 < NCHUNK) {
            const float* pf = mrow + (size_t)(lane >> 1) * T_SZ
                                   + (k0 + KC) + (lane & 1) * 32;
            __builtin_prefetch(pf, 0, 1);
        }

        // ---- S = Q * K^T : four 16x16 C fragments (keys 16j..16j+15) ----
        floatx8 c[4];
        #pragma unroll
        for (int j = 0; j < 4; ++j) {
            floatx8 acc;
            #pragma unroll
            for (int e = 0; e < 8; ++e) acc[e] = 0.0f;
            #pragma unroll
            for (int s = 0; s < 2; ++s) {
                // B fragment: lane column = key 16j+ln, K(d) = 32s + {8h..8h+7, 16+8h..16+8h+7}
                const unsigned short* bp = &sK[(16 * j + ln) * D_SZ + 32 * s + 8 * h];
                bf16x16 bK = ld_frag(bp, bp + 16);
                acc = __builtin_amdgcn_wmma_f32_16x16x32_bf16(
                          false, aQ[s], false, bK, (short)0, acc, false, false);
            }
            c[j] = acc;
        }

        // ---- scale + mask (fp32) ----
        #pragma unroll
        for (int j = 0; j < 4; ++j)
            #pragma unroll
            for (int r = 0; r < 8; ++r)
                c[j][r] = c[j][r] * scale +
                          mrow[(size_t)(r + 8 * h) * T_SZ + k0 + 16 * j + ln];

        // ---- online softmax: row max (ds_swizzle xor-reduce), alpha rescale ----
        float al[8];
        #pragma unroll
        for (int r = 0; r < 8; ++r) {
            float t0 = fmaxf(fmaxf(c[0][r], c[1][r]), fmaxf(c[2][r], c[3][r]));
            SWZ_MAXF(t0, 0x041F);  // xor 1
            SWZ_MAXF(t0, 0x081F);  // xor 2
            SWZ_MAXF(t0, 0x101F);  // xor 4
            SWZ_MAXF(t0, 0x201F);  // xor 8
            const float mnew = fmaxf(m_r[r], t0);
            al[r] = __expf(m_r[r] - mnew);
            m_r[r] = mnew;
        }
        #pragma unroll
        for (int r = 0; r < 8; ++r) {
            #pragma unroll
            for (int j = 0; j < 4; ++j) c[j][r] = __expf(c[j][r] - m_r[r]);
            #pragma unroll
            for (int t = 0; t < 4; ++t) o[t][r] *= al[r];
            oL[r] *= al[r];
        }

        // ---- stage P (bf16) via wave-private LDS, reload in A-fragment layout ----
        unsigned short* Pt = sP[wave];
        #pragma unroll
        for (int r = 0; r < 8; ++r) {
            const int row = (r + 8 * h) * KC;
            #pragma unroll
            for (int j = 0; j < 4; ++j)
                Pt[row + 16 * j + ln] =
                    (unsigned short)((__float_as_uint(c[j][r]) + 0x8000u) >> 16);
        }
        asm volatile("" ::: "memory");  // DS ops are in-order within a wave

        #pragma unroll
        for (int s = 0; s < 2; ++s) {
            // A fragment: row = ln, keys 32s + {8h..8h+7, 16+8h..16+8h+7}
            bf16x16 aP = ld_frag(&Pt[ln * KC + 32 * s + 8 * h],
                                 &Pt[ln * KC + 32 * s + 16 + 8 * h]);
            // row sums: P x ones  (softmax denominator, same bf16 P as PV)
            oL = __builtin_amdgcn_wmma_f32_16x16x32_bf16(
                     false, aP, false, bOnes, (short)0, oL, false, false);
            // O += P x V (4 dim-tiles of 16)
            #pragma unroll
            for (int t = 0; t < 4; ++t) {
                const unsigned short* bp = &sVT[(16 * t + ln) * KC + 32 * s + 8 * h];
                bf16x16 bV = ld_frag(bp, bp + 16);
                o[t] = __builtin_amdgcn_wmma_f32_16x16x32_bf16(
                           false, aP, false, bV, (short)0, o[t], false, false);
            }
        }
    }

    // ---- normalize and write out ----
    #pragma unroll
    for (int r = 0; r < 8; ++r) {
        const float inv = 1.0f / oL[r];
        const size_t row = (size_t)batch * T_SZ + q0 + r + 8 * h;
        #pragma unroll
        for (int t = 0; t < 4; ++t)
            out[row * D_SZ + 16 * t + ln] = o[t][r] * inv;
    }
}

extern "C" void kernel_launch(void* const* d_in, const int* in_sizes, int n_in,
                              void* d_out, int out_size, void* d_ws, size_t ws_size,
                              hipStream_t stream) {
    const float* Q    = (const float*)d_in[0];
    const float* K    = (const float*)d_in[1];
    const float* V    = (const float*)d_in[2];
    const float* mask = (const float*)d_in[3];
    float* out = (float*)d_out;

    const int grid = B_SZ * (T_SZ / QBLK);   // 4 * 64 = 256 blocks
    fa_bf16_wmma_kernel<<<grid, WAVES * 32, 0, stream>>>(Q, K, V, mask, out);
}
